// ContinuousFilterConvolution_20332375179742
// MI455X (gfx1250) — matrix operand order repaired and checked by reading, not compiled, not run on hardware
//
#include <hip/hip_runtime.h>
#include <hip/hip_bf16.h>

typedef __attribute__((ext_vector_type(16))) _Float16 v16h;
typedef __attribute__((ext_vector_type(8)))  _Float16 v8h;
typedef __attribute__((ext_vector_type(8)))  float    v8f;

#define DCH        128   // feature dim D
#define NRBF       64
#define W1T_STRIDE 72    // halves per row (64 K + pad), 144B row, 16B aligned
#define W2T_STRIDE 136   // halves per row (128 K + pad), 272B row, 16B aligned
#define STG_STRIDE 136   // staging row stride (halves)
#define WAVES      8     // waves per block
#define TILE_E     32    // edges per wave (2 WMMA M-tiles)

// LDS layout (bytes):
//   [0, 18432)           W1^T f16 [128 cols][72]
//   [18432, 53248)       W2^T f16 [128 cols][136]
//   [53248, 53760)       cg: interleaved {center[k], -gamma[k]*log2e} f32 pairs
//   [53760, 123392)      staging: WAVES * 32 rows * 136 halves
#define LDS_W_BYTES   (2 * (128 * W1T_STRIDE + 128 * W2T_STRIDE))
#define LDS_TOTAL     (LDS_W_BYTES + 512 + WAVES * TILE_E * STG_STRIDE * 2)

#define LN2f   0.69314718055994531f
#define LOG2Ef 1.4426950408889634f

// shifted softplus, select-free: clamp at 60 keeps exp2 finite and the
// output saturates at softplus(60) - ln2 for x >= 60, not x - ln2.
__device__ __forceinline__ float sspc(float x) {
  float e = __builtin_amdgcn_exp2f(fminf(x, 60.0f) * LOG2Ef);
  return __builtin_amdgcn_logf(1.0f + e) * LN2f - LN2f;   // mul+sub -> fma
}

// Build a 16-half operand from two 8-half (16B-aligned) LDS chunks.
__device__ __forceinline__ v16h ld2x8(const _Float16* p, int gapHalves) {
  v8h lo = *(const v8h*)p;
  v8h hi = *(const v8h*)(p + gapHalves);
  return __builtin_shufflevector(lo, hi, 0,1,2,3,4,5,6,7,8,9,10,11,12,13,14,15);
}

__global__ void cfconv_zero(float* __restrict__ o, long long n) {
  long long i = (long long)blockIdx.x * blockDim.x + threadIdx.x;
  long long stride = (long long)gridDim.x * blockDim.x;
  for (; i < n; i += stride) o[i] = 0.0f;
}

// Convert W1 [64][128] -> W1^T f16 [128][72], W2 [128][128] -> W2^T f16 [128][136]
__global__ void cfconv_prep(const float* __restrict__ W1, const float* __restrict__ W2,
                            _Float16* __restrict__ wsW) {
  _Float16* w1t = wsW;
  _Float16* w2t = wsW + 128 * W1T_STRIDE;
  const int stride = gridDim.x * blockDim.x;
  const int tid = blockIdx.x * blockDim.x + threadIdx.x;
  for (int i = tid; i < 128 * W1T_STRIDE; i += stride) {
    int c = i / W1T_STRIDE, k = i % W1T_STRIDE;
    w1t[i] = (_Float16)((k < NRBF) ? W1[k * DCH + c] : 0.0f);
  }
  for (int i = tid; i < 128 * W2T_STRIDE; i += stride) {
    int c = i / W2T_STRIDE, k = i % W2T_STRIDE;
    w2t[i] = (_Float16)((k < DCH) ? W2[k * DCH + c] : 0.0f);
  }
}

template <bool GUARD>
__device__ __forceinline__ void
wave_body(const float* __restrict__ af,      // atom_features for this batch
          float*       __restrict__ ob,      // output for this batch
          const float* __restrict__ distB,   // distances for this batch
          const int*   __restrict__ idx_j,
          const int*   __restrict__ seg_i,
          const float* __restrict__ b1,
          const float* __restrict__ b2,
          const _Float16* __restrict__ w1t,
          const _Float16* __restrict__ w2t,
          const float* __restrict__ cg,
          _Float16* __restrict__ stg,
          int e0, int col, int hi, int E) {
  // ---- RBF expansion into 16x32 f16 A-operand layout, 2 M-tiles x 2 K-tiles ----
  float dist[2];
  #pragma unroll
  for (int u = 0; u < 2; ++u) {
    int e = e0 + 16 * u + col;
    dist[u] = (!GUARD || e < E) ? distB[e] : 0.0f;
  }
  v16h A1[2][2];
  #pragma unroll
  for (int t = 0; t < 2; ++t) {
    #pragma unroll
    for (int h = 0; h < 16; h += 2) {
      // A layout: lane(0-15) row=col, K = 32t + 8*hi + (h<8 ? h : h+8); h even -> k even
      int k = 32 * t + 8 * hi + (h < 8 ? h : h + 8);
      float4 p = *(const float4*)(cg + 2 * k);          // {c0, g0, c1, g1}
      #pragma unroll
      for (int u = 0; u < 2; ++u) {
        float x0 = dist[u] - p.x;
        float x1 = dist[u] - p.z;
        A1[u][t][h]     = (_Float16)__builtin_amdgcn_exp2f(p.y * (x0 * x0));
        A1[u][t][h + 1] = (_Float16)__builtin_amdgcn_exp2f(p.w * (x1 * x1));
      }
    }
  }

  // ---- GEMM1: [16x64]x[64x128] twice, shared B operands -> ssp -> stage ----
  #pragma unroll
  for (int n = 0; n < 8; ++n) {
    const _Float16* bp = w1t + (n * 16 + col) * W1T_STRIDE;
    v8f cc[2] = {{}, {}};
    #pragma unroll
    for (int t = 0; t < 2; ++t) {
      v16h Bv = ld2x8(bp + 32 * t + 8 * hi, 16);   // B: K at 32t+8hi+0..7 and 32t+16+8hi+0..7
      cc[0] = __builtin_amdgcn_wmma_f32_16x16x32_f16(false, A1[0][t], false, Bv,
                                                     (short)0, cc[0], false, false);
      cc[1] = __builtin_amdgcn_wmma_f32_16x16x32_f16(false, A1[1][t], false, Bv,
                                                     (short)0, cc[1], false, false);
    }
    int ch = n * 16 + col;
    float bias = b1[ch];
    #pragma unroll
    for (int u = 0; u < 2; ++u) {
      #pragma unroll
      for (int r = 0; r < 8; ++r) {
        int m = 16 * u + 8 * hi + r;          // C layout row
        stg[m * STG_STRIDE + ch] = (_Float16)sspc(cc[u][r] + bias);
      }
    }
  }

  // ---- reload filters as A operands for GEMM2 (2 M-tiles x 4 K-tiles) ----
  v16h A2[2][4];
  #pragma unroll
  for (int u = 0; u < 2; ++u) {
    #pragma unroll
    for (int t = 0; t < 4; ++t) {
      const _Float16* p = stg + (16 * u + col) * STG_STRIDE + 32 * t + 8 * hi;
      A2[u][t] = ld2x8(p, 16);   // halves 0-7: k0..k0+7 ; 8-15: k0+16..k0+23
    }
  }

  // per-row edge indices + segment-run flush flags (uniform per half-wave)
  int jj[2][8], ii[2][8];
  bool fl[2][8];
  #pragma unroll
  for (int u = 0; u < 2; ++u) {
    #pragma unroll
    for (int r = 0; r < 8; ++r) {
      int e = e0 + 16 * u + 8 * hi + r;
      bool ok = !GUARD || (e < E);
      jj[u][r] = ok ? idx_j[e] : 0;
      ii[u][r] = ok ? seg_i[e] : 0;
    }
    #pragma unroll
    for (int r = 0; r < 7; ++r) fl[u][r] = (ii[u][r + 1] != ii[u][r]);
    fl[u][7] = true;
  }

  // ---- GEMM2: [16x128]x[128x128] twice with shared B, fused ssp+gather+RLE atomics ----
  #pragma unroll
  for (int n = 0; n < 8; ++n) {
    const _Float16* bp = w2t + (n * 16 + col) * W2T_STRIDE;
    v8f cc[2] = {{}, {}};
    #pragma unroll
    for (int t = 0; t < 4; ++t) {
      v16h Bv = ld2x8(bp + 32 * t + 8 * hi, 16);
      cc[0] = __builtin_amdgcn_wmma_f32_16x16x32_f16(false, A2[0][t], false, Bv,
                                                     (short)0, cc[0], false, false);
      cc[1] = __builtin_amdgcn_wmma_f32_16x16x32_f16(false, A2[1][t], false, Bv,
                                                     (short)0, cc[1], false, false);
    }
    int ch = n * 16 + col;
    float bias = b2[ch];
    #pragma unroll
    for (int u = 0; u < 2; ++u) {
      float acc = 0.0f;
      #pragma unroll
      for (int r = 0; r < 8; ++r) {
        float f = sspc(cc[u][r] + bias);
        float neigh = af[jj[u][r] * DCH + ch];
        float msg = neigh * f;
        if (GUARD) {
          int e = e0 + 16 * u + 8 * hi + r;
          msg = (e < E) ? msg : 0.0f;
        }
        acc += msg;
        if (fl[u][r]) {   // segment boundary: flush run with one atomic
          __hip_atomic_fetch_add(&ob[ii[u][r] * DCH + ch], acc,
                                 __ATOMIC_RELAXED, __HIP_MEMORY_SCOPE_AGENT);
          acc = 0.0f;
        }
      }
    }
  }
}

__global__ void __launch_bounds__(32 * WAVES)
cfconv_main(const float* __restrict__ atom_features,
            const float* __restrict__ distances,
            const int*   __restrict__ idx_j,
            const int*   __restrict__ seg_i,
            const float* __restrict__ centers,
            const float* __restrict__ gamma,
            const float* __restrict__ b1,
            const float* __restrict__ b2,
            const _Float16* __restrict__ wsW,
            float* __restrict__ out,
            int N, int E) {
  extern __shared__ char smem[];
  _Float16* w1t  = (_Float16*)smem;
  _Float16* w2t  = w1t + 128 * W1T_STRIDE;
  float*    cg   = (float*)(smem + LDS_W_BYTES);   // interleaved {center, -gamma*log2e}
  _Float16* stageBase = (_Float16*)(smem + LDS_W_BYTES + 512);

  const int tid = threadIdx.x;
  // cooperative copy of W tiles + interleaved rbf params into LDS
  {
    const unsigned* src = (const unsigned*)wsW;
    unsigned* dst = (unsigned*)smem;
    const int nWords = LDS_W_BYTES / 4;
    for (int i = tid; i < nWords; i += blockDim.x) dst[i] = src[i];
    for (int i = tid; i < NRBF; i += blockDim.x) {
      cg[2 * i]     = centers[i];
      cg[2 * i + 1] = -LOG2Ef * gamma[i];
    }
  }
  __syncthreads();

  const int lane = tid & 31;
  const int wave = tid >> 5;
  const int col  = lane & 15;
  const int hi   = lane >> 4;
  const int b    = blockIdx.y;
  const int e0   = (blockIdx.x * WAVES + wave) * TILE_E;
  if (e0 >= E) return;

  _Float16* stg = stageBase + wave * TILE_E * STG_STRIDE;
  const float* af    = atom_features + (long long)b * N * DCH;
  float*       ob    = out           + (long long)b * N * DCH;
  const float* distB = distances     + (long long)b * E;

  if (e0 + TILE_E <= E) {
    wave_body<false>(af, ob, distB, idx_j, seg_i, b1, b2, w1t, w2t, cg, stg,
                     e0, col, hi, E);
  } else {
    wave_body<true>(af, ob, distB, idx_j, seg_i, b1, b2, w1t, w2t, cg, stg,
                    e0, col, hi, E);
  }
}

extern "C" void kernel_launch(void* const* d_in, const int* in_sizes, int n_in,
                              void* d_out, int out_size, void* d_ws, size_t ws_size,
                              hipStream_t stream) {
  const float* atom_features = (const float*)d_in[0];
  const float* distances     = (const float*)d_in[1];
  const int*   idx_j         = (const int*)d_in[2];
  const int*   seg_i         = (const int*)d_in[3];
  const float* centers       = (const float*)d_in[4];
  const float* gamma         = (const float*)d_in[5];
  const float* W1            = (const float*)d_in[6];
  const float* b1            = (const float*)d_in[7];
  const float* W2            = (const float*)d_in[8];
  const float* b2            = (const float*)d_in[9];
  float* out = (float*)d_out;

  const int E = in_sizes[2];
  const int B = in_sizes[1] / E;
  const int N = (int)(in_sizes[0] / ((long long)B * DCH));

  // 1) zero accumulation output (harness poisons d_out)
  {
    long long n = out_size;
    int blocks = (int)((n + 255) / 256);
    if (blocks > 4096) blocks = 4096;
    cfconv_zero<<<blocks, 256, 0, stream>>>(out, n);
  }
  // 2) pre-transpose + f16-convert W1/W2 into workspace
  _Float16* wsW = (_Float16*)d_ws;
  cfconv_prep<<<64, 256, 0, stream>>>(W1, W2, wsW);

  // 3) fused RBF -> WMMA MLP -> gather/scatter
  int tiles = (E + TILE_E - 1) / TILE_E;
  int gridX = (tiles + WAVES - 1) / WAVES;
  dim3 grid(gridX, B);
  cfconv_main<<<grid, 32 * WAVES, LDS_TOTAL, stream>>>(
      atom_features, distances, idx_j, seg_i, centers, gamma,
      b1, b2, wsW, out, N, E);
}
